// SO2MLP_88656714925180
// MI455X (gfx1250) — matrix operand order repaired and checked by reading, not compiled, not run on hardware
//
#include <hip/hip_runtime.h>

#define N_NODES 20000
#define L_MAX   6
#define R_IN    256
#define R_HID   512
#define R_OUT   256

typedef __attribute__((ext_vector_type(16))) _Float16 v16h;
typedef __attribute__((ext_vector_type(8)))  _Float16 v8h;
typedef __attribute__((ext_vector_type(8)))  float    v8f;

// ---------------------------------------------------------------------------
// Pre-pass: convert both weight tensors to f16 in workspace.
// W1: (L, R_HID, R_IN) row-major; W2: (L, R_OUT, R_HID) row-major.
// ---------------------------------------------------------------------------
__global__ void so2_cvt_weights(const float* __restrict__ W1,
                                const float* __restrict__ W2,
                                _Float16* __restrict__ w1h,
                                _Float16* __restrict__ w2h) {
  int idx = blockIdx.x * blockDim.x + threadIdx.x;
  if (idx < L_MAX * R_HID * R_IN) {
    w1h[idx] = (_Float16)W1[idx];
    w2h[idx] = (_Float16)W2[idx];
  }
}

// ---------------------------------------------------------------------------
// Fused SO2-MLP: layer1 (WMMA) -> rotate/leaky/rotate-back (VALU on C tiles)
// -> LDS relayout -> layer2 (WMMA) -> scatter store.
// Block = 8 nodes (16 WMMA rows), 6 waves, wave w handles degree l = w.
// Row ordering M = 2*node_local + comp  ==> the two components of one node
// land in adjacent VGPRs of the same lane in the 16x16 f32 C/D layout.
// ---------------------------------------------------------------------------
__launch_bounds__(192)
__global__ void so2mlp_fused(const float* __restrict__ x,     // (n, R_IN, L*2)
                             const float* __restrict__ rot,   // (n, L, 2, 2)
                             const _Float16* __restrict__ w1h,
                             const _Float16* __restrict__ w2h,
                             float* __restrict__ out) {       // (n, R_OUT, L*2)
  __shared__ __align__(16) _Float16 hbuf[L_MAX][16][R_HID];   // 96 KB

  const int lane  = threadIdx.x & 31;
  const int l     = threadIdx.x >> 5;   // wave id == degree index
  const int hsel  = lane >> 4;          // lane half (A/B K-interleave select)
  const int m     = lane & 15;          // A row / B column within tile
  const int node0 = blockIdx.x * 8;

  // --- rotation coefficients: VGPR pair p holds rows M=2p(+8h),2p+1(+8h)
  //     => node_local = p + 4*hsel for this lane ---
  float cp[4], sp[4];
  #pragma unroll
  for (int p = 0; p < 4; ++p) {
    int n = node0 + p + 4 * hsel;
    const float* rp = rot + (size_t)((n * L_MAX + l) * 2) * 2;
    cp[p] = rp[0];   // rot[n][l][0][0] =  cos
    sp[p] = rp[2];   // rot[n][l][1][0] =  sin
  }

  // --- layer-1 A fragments (16x256 f16 tile, rows = (node,comp)) ---
  // A fragment (contiguous K order used here): lane row M = lane%16; elems e -> K = 16*hsel + e,
  // per 32-wide K step (matches B).
  const int node = node0 + (m >> 1);
  const int comp = m & 1;
  const float* xr = x + (size_t)node * (R_IN * L_MAX * 2) + (l * 2 + comp);

  v16h a1[8];
  #pragma unroll
  for (int ks = 0; ks < 8; ++ks) {
    v16h f;
    #pragma unroll
    for (int e = 0; e < 8; ++e) {
      int k0 = ks * 32 + hsel * 16 + e;
      int k1 = ks * 32 + hsel * 16 + 8 + e;
      f[e]     = (_Float16)xr[(size_t)k0 * (L_MAX * 2)];
      f[e + 8] = (_Float16)xr[(size_t)k1 * (L_MAX * 2)];
    }
    a1[ks] = f;
  }

  // --- layer 1 + rotation activation, streamed per 16-wide r_hid tile ---
  // B fragment (same contiguous K order as A): lane col N = lane%16, elems e -> K = 16*hsel + e
  // => 32 contiguous bytes of W[l][o][:] per lane per K-step.
  const _Float16* w1l = w1h + (size_t)l * R_HID * R_IN;
  for (int nt = 0; nt < R_HID / 16; ++nt) {
    v8f acc = {};
    const _Float16* bp = w1l + (size_t)(nt * 16 + m) * R_IN + hsel * 16;
    #pragma unroll
    for (int ks = 0; ks < 8; ++ks) {
      v16h b = *(const v16h*)(bp + ks * 32);
      acc = __builtin_amdgcn_wmma_f32_16x16x32_f16(false, a1[ks], false, b,
                                                   (short)0, acc, false, false);
    }
    #pragma unroll
    for (int p = 0; p < 4; ++p) {
      float h0 = acc[2 * p], h1 = acc[2 * p + 1];
      float c = cp[p], s = sp[p];
      float a0 = h0 * c + h1 * s;          // into local frame
      float b1 = h1 * c - h0 * s;
      a0 = (a0 >= 0.f) ? a0 : 0.01f * a0;  // leaky_relu
      b1 = (b1 >= 0.f) ? b1 : 0.01f * b1;
      float o0 = a0 * c - b1 * s;          // back to global frame
      float o1 = a0 * s + b1 * c;
      int col = nt * 16 + m;
      hbuf[l][2 * p +     8 * hsel][col] = (_Float16)o0;
      hbuf[l][2 * p + 1 + 8 * hsel][col] = (_Float16)o1;
    }
  }

  // --- relayout h: C-layout (LDS, row-major 16x512) -> layer-2 A fragments ---
  v16h a2[16];
  #pragma unroll
  for (int ks = 0; ks < 16; ++ks) {
    const _Float16* p0 = &hbuf[l][m][ks * 32 + hsel * 16];        // ds_load_b128
    const _Float16* p1 = &hbuf[l][m][ks * 32 + hsel * 16 + 8];   // ds_load_b128
    v8h lo = *(const v8h*)p0;
    v8h hi = *(const v8h*)p1;
    a2[ks] = __builtin_shufflevector(lo, hi,
                                     0,1,2,3,4,5,6,7,8,9,10,11,12,13,14,15);
  }

  // --- layer 2: out(16 x 256), K = 512 ---
  const _Float16* w2l = w2h + (size_t)l * R_OUT * R_HID;
  for (int nt = 0; nt < R_OUT / 16; ++nt) {
    v8f acc = {};
    const _Float16* bp = w2l + (size_t)(nt * 16 + m) * R_HID + hsel * 16;
    #pragma unroll
    for (int ks = 0; ks < 16; ++ks) {
      v16h b = *(const v16h*)(bp + ks * 32);
      acc = __builtin_amdgcn_wmma_f32_16x16x32_f16(false, a2[ks], false, b,
                                                   (short)0, acc, false, false);
    }
    #pragma unroll
    for (int r = 0; r < 8; ++r) {
      int M  = r + 8 * hsel;
      int n  = node0 + (M >> 1);
      int cc = M & 1;
      int o  = nt * 16 + m;
      out[((size_t)n * R_OUT + o) * (L_MAX * 2) + l * 2 + cc] = acc[r];
    }
  }
}

// ---------------------------------------------------------------------------
extern "C" void kernel_launch(void* const* d_in, const int* in_sizes, int n_in,
                              void* d_out, int out_size, void* d_ws, size_t ws_size,
                              hipStream_t stream) {
  const float* x   = (const float*)d_in[0];   // (20000, 256, 12) f32
  const float* rot = (const float*)d_in[1];   // (20000, 6, 2, 2) f32
  const float* W1  = (const float*)d_in[2];   // (6, 512, 256) f32
  const float* W2  = (const float*)d_in[3];   // (6, 256, 512) f32

  _Float16* w1h = (_Float16*)d_ws;                       // 1.5 MB
  _Float16* w2h = w1h + (size_t)L_MAX * R_HID * R_IN;    // 1.5 MB

  const int nw = L_MAX * R_HID * R_IN;
  so2_cvt_weights<<<(nw + 255) / 256, 256, 0, stream>>>(W1, W2, w1h, w2h);

  so2mlp_fused<<<N_NODES / 8, 192, 0, stream>>>(x, rot, w1h, w2h, (float*)d_out);
}
